// ScaledDotProductAttention_40733469835925
// MI455X (gfx1250) — compile-verified, matrix operand order checked by reading, not run on hardware
//
#include <hip/hip_runtime.h>

typedef __attribute__((ext_vector_type(16))) _Float16 v16h;
typedef __attribute__((ext_vector_type(8)))  float    v8f;
typedef __attribute__((ext_vector_type(4)))  float    f32x4;

constexpr int Bn = 2, Hn = 16, Sn = 2048, Dn = 64;
constexpr int QT = Sn / 16;          // 128 q-tiles per (b,h)
constexpr int SD = 2052;             // padded LDS row stride in floats (2048+4 -> conflict-free)
constexpr float SCALE = 0.125f;      // 64^-0.5

// Build a 16-bit A/B WMMA fragment from a contiguous fp32 row.
// CDNA5 16-bit 16x32 A layout: lane holds K = {base1..base1+7} in halves 0..7
// and K = {base2..base2+7} in halves 8..15 (base1 = 32*chunk + 8*grp, base2 = base1+16).
__device__ __forceinline__ v16h load_frag_row(const float* __restrict__ row,
                                              int base1, int base2) {
  v16h r;
#pragma unroll
  for (int i = 0; i < 8; ++i) {
    r[i]     = (_Float16)row[base1 + i];
    r[8 + i] = (_Float16)row[base2 + i];
  }
  return r;
}

__global__ __launch_bounds__(128)
void fattn_fwd_kernel(const float* __restrict__ q,
                      const float* __restrict__ kmat,
                      const float* __restrict__ vmat,
                      float* __restrict__ out_o,
                      float* __restrict__ out_w) {
  extern __shared__ float smem[];
  float* sc     = smem;             // [16][SD]: scores, then exp(s - rowmax)
  float* rowmax = smem + 16 * SD;   // [16]
  float* rowsum = rowmax + 16;      // [16]

  const int tid  = threadIdx.x;
  const int wave = tid >> 5;
  const int lane = tid & 31;
  const int grp  = lane >> 4;       // lane half (0/1)
  const int l16  = lane & 15;

  const int bh    = blockIdx.x >> 7;     // (b*H + h)
  const int qt    = blockIdx.x & (QT - 1);
  const int q0    = qt * 16;
  const int ncols = (qt + 1) * 16;       // causal column extent for this q-tile

  const float* Q = q    + (size_t)bh * Sn * Dn;
  const float* K = kmat + (size_t)bh * Sn * Dn;
  const float* V = vmat + (size_t)bh * Sn * Dn;

  // ---- Preload Q A-fragments (two 32-wide K chunks cover D = 64) --------
  v16h aq[2];
  {
    const float* qrow = Q + (size_t)(q0 + l16) * Dn;
#pragma unroll
    for (int ch = 0; ch < 2; ++ch)
      aq[ch] = load_frag_row(qrow, 32 * ch + 8 * grp, 32 * ch + 16 + 8 * grp);
  }

  // ---- Phase 1: S = (Q K^T) * SCALE with causal mask -> LDS -------------
  // 4 waves split the k-tiles of this q-tile's causal range.
  for (int kt = wave; kt <= qt; kt += 4) {
    const float* krow = K + (size_t)(kt * 16 + l16) * Dn;
    v8f c = {};
#pragma unroll
    for (int ch = 0; ch < 2; ++ch) {
      v16h bk = load_frag_row(krow, 32 * ch + 8 * grp, 32 * ch + 16 + 8 * grp);
      c = __builtin_amdgcn_wmma_f32_16x16x32_f16(false, aq[ch], false, bk,
                                                 (short)0, c, false, false);
    }
    // C/D layout: lane half 'grp', VGPR vv -> M = vv + 8*grp, N = l16
#pragma unroll
    for (int vv = 0; vv < 8; ++vv) {
      const int m = vv + 8 * grp;
      float s = c[vv] * SCALE;
      if (kt == qt && l16 > m) s = -__builtin_inff();  // causal mask on diagonal tile
      sc[m * SD + kt * 16 + l16] = s;
    }
  }
  __syncthreads();

  // ---- Phase 2: exact softmax stats in LDS ------------------------------
  const int r  = tid >> 3;   // 8 lanes per row
  const int jj = tid & 7;
  {
    float mx = -__builtin_inff();
    for (int col = jj; col < ncols; col += 8)
      mx = fmaxf(mx, sc[r * SD + col]);
#pragma unroll
    for (int d = 1; d < 8; d <<= 1)
      mx = fmaxf(mx, __shfl_xor(mx, d, 32));
    if (jj == 0) rowmax[r] = mx;
  }
  __syncthreads();
  {
    const float mx = rowmax[r];
    float sum = 0.f;
    for (int col = jj; col < ncols; col += 8) {
      float p = __expf(sc[r * SD + col] - mx);
      sc[r * SD + col] = p;        // keep unnormalized probabilities in LDS
      sum += p;
    }
#pragma unroll
    for (int d = 1; d < 8; d <<= 1)
      sum += __shfl_xor(sum, d, 32);
    if (jj == 0) rowsum[r] = sum;
  }
  __syncthreads();

  // ---- Phase 3: stream normalized weights (and causal zeros) to HBM -----
  {
    float* wbase = out_w + (size_t)bh * Sn * Sn + (size_t)q0 * Sn;
    for (int idx = tid; idx < 16 * (Sn / 4); idx += 128) {
      const int row = idx >> 9;          // Sn/4 = 512 float4 per row
      const int col = (idx & 511) * 4;   // ncols is a multiple of 16 -> vec-aligned
      f32x4 val;
      if (col < ncols) {
        const float inv = 1.0f / rowsum[row];
        const float* p = &sc[row * SD + col];
        val.x = p[0] * inv; val.y = p[1] * inv;
        val.z = p[2] * inv; val.w = p[3] * inv;
      } else {
        val = (f32x4){0.f, 0.f, 0.f, 0.f};
      }
      *(f32x4*)(wbase + (size_t)row * Sn + col) = val;
    }
  }

  // ---- Phase 4: O = P V via WMMA; each wave owns 16 d-columns -----------
  {
    const int dc  = wave * 16 + l16;
    const int nch = (ncols + 31) >> 5;   // 32-wide K chunks
    v8f o = {};
    for (int ch = 0; ch < nch; ++ch) {
      const int base1 = 32 * ch + 8 * grp;
      const int base2 = base1 + 16;
      // A: unnormalized probabilities from LDS (rows m = l16)
      v16h pa;
      const float* prow = sc + (size_t)l16 * SD;
#pragma unroll
      for (int i = 0; i < 8; ++i) {
        const int c1 = base1 + i, c2 = base2 + i;
        pa[i]     = (_Float16)(c1 < ncols ? prow[c1] : 0.f);
        pa[8 + i] = (_Float16)(c2 < ncols ? prow[c2] : 0.f);
      }
      // B: V rows (k) x columns (dc); V is L2-resident (48 MB << 192 MB L2)
      v16h vb;
#pragma unroll
      for (int i = 0; i < 8; ++i) {
        const int k1 = base1 + i, k2 = base2 + i;
        vb[i]     = (_Float16)(k1 < ncols ? V[(size_t)k1 * Dn + dc] : 0.f);
        vb[8 + i] = (_Float16)(k2 < ncols ? V[(size_t)k2 * Dn + dc] : 0.f);
      }
      o = __builtin_amdgcn_wmma_f32_16x16x32_f16(false, pa, false, vb,
                                                 (short)0, o, false, false);
    }
    float* orow = out_o + (size_t)bh * Sn * Dn + (size_t)q0 * Dn;
#pragma unroll
    for (int vv = 0; vv < 8; ++vv) {
      const int m = vv + 8 * grp;
      orow[(size_t)m * Dn + dc] = o[vv] * (1.0f / rowsum[m]);
    }
  }
}

extern "C" void kernel_launch(void* const* d_in, const int* in_sizes, int n_in,
                              void* d_out, int out_size, void* d_ws, size_t ws_size,
                              hipStream_t stream) {
  (void)in_sizes; (void)n_in; (void)out_size; (void)d_ws; (void)ws_size;
  const float* q = (const float*)d_in[0];
  const float* k = (const float*)d_in[1];
  const float* v = (const float*)d_in[2];
  // d_in[3] is the causal mask; causality is computed analytically in-kernel.

  float* out_o = (float*)d_out;                               // [B,H,S,D]
  float* out_w = out_o + (size_t)Bn * Hn * Sn * Dn;           // [B,H,S,S]

  const dim3 grid(Bn * Hn * QT);   // 4096 workgroups, one per (b,h,q-tile)
  const dim3 block(128);           // 4 waves (wave32)
  const size_t lds_bytes = (size_t)(16 * SD + 32) * sizeof(float);  // ~131.5 KB

  fattn_fwd_kernel<<<grid, block, lds_bytes, stream>>>(q, k, v, out_o, out_w);
}
